// NodeBlock_77524159693412
// MI455X (gfx1250) — matrix operand order repaired and checked by reading, not compiled, not run on hardware
//
#include <hip/hip_runtime.h>
#include <hip/hip_bf16.h>

typedef __bf16 bf16_t;
typedef __attribute__((ext_vector_type(16))) __bf16 bf16x16;
typedef __attribute__((ext_vector_type(8)))  __bf16 bf16x8;
typedef __attribute__((ext_vector_type(8)))  float  f32x8;
typedef __attribute__((ext_vector_type(4)))  float  f32x4;

#define NN 50000
#define EE 1600000
#define DE_ 128
#define DN_ 128
#define DG_ 128
#define DIN_ 384
#define DH_ 128
#define DO_ 128
#define WAVES 2
#define W_ELEMS (DH_ * DIN_ + DO_ * DH_)   // 65536 bf16 = 131072 bytes

// ---------------- kernel 1: zero accumulator + counts ----------------
__global__ __launch_bounds__(256) void k_init(float* __restrict__ agg,
                                              float* __restrict__ cnt) {
  size_t i = (size_t)blockIdx.x * blockDim.x + threadIdx.x;
  size_t na = (size_t)NN * DE_;
  if (i < na)            agg[i] = 0.0f;
  else if (i < na + NN)  cnt[i - na] = 0.0f;
}

// ---------------- kernel 2: transpose+convert weights to bf16 ----------------
__global__ __launch_bounds__(256) void k_convw(const float* __restrict__ W1,
                                               const float* __restrict__ W2,
                                               bf16_t* __restrict__ w1t,
                                               bf16_t* __restrict__ w2t) {
  int i = blockIdx.x * blockDim.x + threadIdx.x;
  if (i < DIN_ * DH_) {                 // W1 [384,128] -> W1T [128,384]
    int k = i / DH_, n = i % DH_;
    w1t[(size_t)n * DIN_ + k] = (bf16_t)W1[i];
  } else {
    int j = i - DIN_ * DH_;
    if (j < DH_ * DO_) {                // W2 [128,128] -> W2T [128,128]
      int k = j / DO_, n = j % DO_;
      w2t[(size_t)n * DH_ + k] = (bf16_t)W2[j];
    }
  }
}

// ---------------- kernel 3: edge scatter (segment sum + counts) ----------------
// Edge stream is one-shot: load non-temporally so the L2 keeps the 25.6MB
// atomic accumulator resident while 819MB streams through.
__global__ __launch_bounds__(256) void k_scatter(const f32x4* __restrict__ ea,
                                                 const int* __restrict__ dst,
                                                 float* __restrict__ agg,
                                                 float* __restrict__ cnt) {
  size_t tid = (size_t)blockIdx.x * blockDim.x + threadIdx.x;
  int e = (int)(tid >> 5);
  int l = (int)(tid & 31);
  if (e >= EE) return;
  if (e + 256 < EE)      // global_prefetch_b8 ahead of the stream
    __builtin_prefetch(ea + ((size_t)(e + 256) * 32 + l), 0, 1);
  int d = dst[e];
  f32x4 v = __builtin_nontemporal_load(&ea[(size_t)e * 32 + l]);
  float* p = agg + (size_t)d * DE_ + l * 4;
  atomicAdd(p + 0, v[0]);
  atomicAdd(p + 1, v[1]);
  atomicAdd(p + 2, v[2]);
  atomicAdd(p + 3, v[3]);
  if (l == 0) atomicAdd(cnt + d, 1.0f);
}

// ---------------- kernel 4: fused mean + concat + 2-layer WMMA MLP ----------------
__global__ __launch_bounds__(WAVES * 32) void k_mlp(const float* __restrict__ agg,
                                                    const float* __restrict__ cnt,
                                                    const float* __restrict__ node_attrs,
                                                    const float* __restrict__ gattr,
                                                    const float* __restrict__ b1,
                                                    const float* __restrict__ b2,
                                                    const bf16_t* __restrict__ w1t,  // w2t contiguous after
                                                    float* __restrict__ out) {
  __shared__ bf16_t wts[W_ELEMS];           // W1T then W2T, bf16   (131072 B)
  __shared__ bf16_t hin[WAVES][16][DIN_];   // staged concat rows   ( 24576 B)
  __shared__ bf16_t h1 [WAVES][16][DH_];    // hidden activations   (  8192 B)

  const int lane = threadIdx.x & 31;
  const int wv   = threadIdx.x >> 5;
  const int tile = blockIdx.x * WAVES + wv;
  const bool active = (tile * 16 < NN);     // wave-uniform
  const int node0 = tile * 16;
  const int half  = lane >> 4;
  const int l16   = lane & 15;

  // ---- async-stage W1T+W2T into LDS (ASYNCcnt pipe, no VGPR round trip) ----
  {
    const uint32_t base = (uint32_t)(uintptr_t)(&wts[0]);
    const uint64_t gsrc = (uint64_t)(uintptr_t)w1t;
    const int nops = (W_ELEMS * 2) / 16;                // 8192 x 16B
#pragma unroll 4
    for (int i = threadIdx.x; i < nops; i += WAVES * 32) {
      uint32_t off    = (uint32_t)i * 16u;
      uint32_t ldsoff = base + off;
      uint64_t ga     = gsrc + off;
      asm volatile("global_load_async_to_lds_b128 %0, %1, off"
                   :: "v"(ldsoff), "v"(ga) : "memory");
    }
  }

  // ---- stage 16 x 384 bf16 input rows (overlaps with async DMA) ----
  if (active) {
    for (int idx = lane; idx < 16 * DIN_; idx += 32) {
      int r = idx / DIN_, c = idx % DIN_;
      int node = node0 + r;
      float v;
      if (c < DE_) {
        float inv = 1.0f / fmaxf(cnt[node], 1.0f);
        v = agg[(size_t)node * DE_ + c] * inv;
      } else if (c < DE_ + DN_) {
        v = node_attrs[(size_t)node * DN_ + (c - DE_)];
      } else {
        v = gattr[c - DE_ - DN_];
      }
      hin[wv][r][c] = (bf16_t)v;
    }
  }

  asm volatile("s_wait_asynccnt 0" ::: "memory");   // my async copies landed
  __syncthreads();                                  // everyone's copies visible
  if (!active) return;

  // ---- layer 1: [16x384] @ [384x128], bias folded into acc init ----
  f32x8 acc[8];
#pragma unroll
  for (int t = 0; t < 8; ++t) {
    float bv = b1[t * 16 + l16];
#pragma unroll
    for (int j = 0; j < 8; ++j) acc[t][j] = bv;
  }
#pragma unroll 1                           // keep k-loop rolled: VGPR pressure
  for (int ks = 0; ks < DIN_ / 32; ++ks) {
    const int kb = ks * 32;
    union { bf16x16 v; bf16x8 h[2]; } A;
    const bf16_t* ap = &hin[wv][l16][kb + (half ? 8 : 0)];
    A.h[0] = *(const bf16x8*)ap;          // K = k0..k0+7
    A.h[1] = *(const bf16x8*)(ap + 16);   // K = k0+16..k0+23
#pragma unroll
    for (int t = 0; t < 8; ++t) {
      union { bf16x16 v; bf16x8 h[2]; } B;
      const bf16_t* bp = &wts[(size_t)(t * 16 + l16) * DIN_ + kb + (half ? 8 : 0)];
      B.h[0] = *(const bf16x8*)bp;
      B.h[1] = *(const bf16x8*)(bp + 16);
      acc[t] = __builtin_amdgcn_wmma_f32_16x16x32_bf16(
          false, A.v, false, B.v, (short)0, acc[t], false, false);
    }
  }

  // ---- ReLU, write hidden tile to LDS in bf16 (C/D layout -> row major) ----
#pragma unroll
  for (int t = 0; t < 8; ++t) {
#pragma unroll
    for (int v = 0; v < 8; ++v) {
      float x = fmaxf(acc[t][v], 0.0f);
      h1[wv][v + (half ? 8 : 0)][t * 16 + l16] = (bf16_t)x;
    }
  }
  asm volatile("s_wait_dscnt 0" ::: "memory");   // intra-wave LDS visibility

  // ---- layer 2: [16x128] @ [128x128], bias folded into acc init ----
  f32x8 acc2[8];
#pragma unroll
  for (int t = 0; t < 8; ++t) {
    float bv = b2[t * 16 + l16];
#pragma unroll
    for (int j = 0; j < 8; ++j) acc2[t][j] = bv;
  }
#pragma unroll 1                           // keep k-loop rolled: VGPR pressure
  for (int ks = 0; ks < DH_ / 32; ++ks) {
    const int kb = ks * 32;
    union { bf16x16 v; bf16x8 h[2]; } A;
    const bf16_t* ap = &h1[wv][l16][kb + (half ? 8 : 0)];
    A.h[0] = *(const bf16x8*)ap;
    A.h[1] = *(const bf16x8*)(ap + 16);
#pragma unroll
    for (int t = 0; t < 8; ++t) {
      union { bf16x16 v; bf16x8 h[2]; } B;
      const bf16_t* bp = &wts[DH_ * DIN_ + (size_t)(t * 16 + l16) * DH_ + kb + (half ? 8 : 0)];
      B.h[0] = *(const bf16x8*)bp;
      B.h[1] = *(const bf16x8*)(bp + 16);
      acc2[t] = __builtin_amdgcn_wmma_f32_16x16x32_bf16(
          false, A.v, false, B.v, (short)0, acc2[t], false, false);
    }
  }

  // ---- store f32 output ----
#pragma unroll
  for (int t = 0; t < 8; ++t) {
#pragma unroll
    for (int v = 0; v < 8; ++v) {
      int row = v + (half ? 8 : 0);
      out[(size_t)(node0 + row) * DO_ + t * 16 + l16] = acc2[t][v];
    }
  }
}

extern "C" void kernel_launch(void* const* d_in, const int* in_sizes, int n_in,
                              void* d_out, int out_size, void* d_ws, size_t ws_size,
                              hipStream_t stream) {
  const float* edge_attrs = (const float*)d_in[0];
  const float* node_attrs = (const float*)d_in[1];
  const float* gattr      = (const float*)d_in[2];
  const float* W1         = (const float*)d_in[3];
  const float* b1         = (const float*)d_in[4];
  const float* W2         = (const float*)d_in[5];
  const float* b2         = (const float*)d_in[6];
  const int*   edge_dst   = (const int*)d_in[7];
  float* out = (float*)d_out;

  // workspace layout (all 16B aligned; w1t and w2t contiguous for async copy)
  float*  agg = (float*)d_ws;                         // N*128 f32
  float*  cnt = agg + (size_t)NN * DE_;               // N f32
  bf16_t* w1t = (bf16_t*)(cnt + NN);                  // 128x384 bf16
  bf16_t* w2t = w1t + (size_t)DH_ * DIN_;             // 128x128 bf16 (contiguous)

  {
    size_t total = (size_t)NN * DE_ + NN;
    int blocks = (int)((total + 255) / 256);
    k_init<<<blocks, 256, 0, stream>>>(agg, cnt);
  }
  {
    int total = DIN_ * DH_ + DH_ * DO_;               // 65536
    k_convw<<<(total + 255) / 256, 256, 0, stream>>>(W1, W2, w1t, w2t);
  }
  {
    size_t total = (size_t)EE * 32;                   // 32 lanes per edge
    int blocks = (int)((total + 255) / 256);
    k_scatter<<<blocks, 256, 0, stream>>>((const f32x4*)edge_attrs, edge_dst, agg, cnt);
  }
  {
    int tiles = NN / 16;                              // 3125
    int blocks = (tiles + WAVES - 1) / WAVES;
    k_mlp<<<blocks, WAVES * 32, 0, stream>>>(agg, cnt, node_attrs, gattr,
                                             b1, b2, w1t, out);
  }
}
